// Block_21569325760810
// MI455X (gfx1250) — compile-verified, matrix operand order checked by reading, not run on hardware
//
#include <hip/hip_runtime.h>

// ---------------- problem constants ----------------
#define B_   2
#define S_   2048
#define D_   1024
#define H_   16
#define HD_  64

typedef _Float16 v8h  __attribute__((ext_vector_type(8)));
typedef _Float16 v16h __attribute__((ext_vector_type(16)));
typedef float    v8f  __attribute__((ext_vector_type(8)));
typedef unsigned u32x4 __attribute__((ext_vector_type(4)));
typedef int      i32x4 __attribute__((ext_vector_type(4)));
typedef int      i32x8 __attribute__((ext_vector_type(8)));

union AFrag { v16h v; v8h h[2]; };

// ---- TDM availability / arity dispatch (probe-verified toolchain difference) ----
#if __has_builtin(__builtin_amdgcn_tensor_load_to_lds) && __has_builtin(__builtin_amdgcn_s_wait_tensorcnt)
  #define HAVE_TDM 1
  #if __has_include(<hip/amd_detail/amd_gfx1250_TDM.h>)
    // amdgpu-toolchain / therock headers: 6-arg form
    #define TDM_LOAD(G0, G1, G2, G3, Z8) \
        __builtin_amdgcn_tensor_load_to_lds((G0), (G1), (G2), (G3), (Z8), 0)
  #else
    // ROCm 7.2 clang-22: 5-arg form
    #define TDM_LOAD(G0, G1, G2, G3, Z8) \
        __builtin_amdgcn_tensor_load_to_lds((G0), (G1), (G2), (G3), 0)
  #endif
#else
  #define HAVE_TDM 0
#endif

__device__ __forceinline__ float gelu_f(float x) {
    return 0.5f * x * (1.0f + tanhf(0.7978845608028654f * (x + 0.044715f * x * x * x)));
}

// ---------------- f32 -> f16 convert ----------------
__global__ void cvt_f32_f16_kernel(const float* __restrict__ in,
                                   _Float16* __restrict__ out, size_t n) {
    size_t i = (size_t)blockIdx.x * blockDim.x + threadIdx.x;
    if (i < n) out[i] = (_Float16)in[i];
}

// ---------------- layernorm (torch-style, unbiased std, /(std+eps)) ----------------
__global__ __launch_bounds__(256)
void layernorm_kernel(const float* __restrict__ x, const float* __restrict__ w,
                      const float* __restrict__ bb, _Float16* __restrict__ out) {
    __shared__ float sbuf[16];
    const int row = blockIdx.x, tid = threadIdx.x;
    const float4 xv = *(const float4*)(x + (size_t)row * D_ + tid * 4);
    float s = xv.x + xv.y + xv.z + xv.w;
    #pragma unroll
    for (int off = 16; off > 0; off >>= 1) s += __shfl_xor(s, off, 32);
    if ((tid & 31) == 0) sbuf[tid >> 5] = s;
    __syncthreads();
    if (tid == 0) { float t = 0.f; for (int i = 0; i < 8; ++i) t += sbuf[i]; sbuf[8] = t; }
    __syncthreads();
    const float mean = sbuf[8] * (1.0f / (float)D_);
    const float cx = xv.x - mean, cy = xv.y - mean, cz = xv.z - mean, cw = xv.w - mean;
    float q = cx * cx + cy * cy + cz * cz + cw * cw;
    #pragma unroll
    for (int off = 16; off > 0; off >>= 1) q += __shfl_xor(q, off, 32);
    __syncthreads();
    if ((tid & 31) == 0) sbuf[tid >> 5] = q;
    __syncthreads();
    if (tid == 0) { float t = 0.f; for (int i = 0; i < 8; ++i) t += sbuf[i]; sbuf[8] = t; }
    __syncthreads();
    const float rstd = 1.0f / (sqrtf(sbuf[8] / (float)(D_ - 1)) + 1e-5f);
    const float4 wv = *(const float4*)(w + tid * 4);
    const float4 bv = *(const float4*)(bb + tid * 4);
    _Float16* op = out + (size_t)row * D_ + tid * 4;
    op[0] = (_Float16)(wv.x * (cx * rstd) + bv.x);
    op[1] = (_Float16)(wv.y * (cy * rstd) + bv.y);
    op[2] = (_Float16)(wv.z * (cz * rstd) + bv.z);
    op[3] = (_Float16)(wv.w * (cw * rstd) + bv.w);
}

// ---------------- tiled WMMA GEMM: C[M,N] = A[M,K](f16) @ B[K,N](f16) + bias ----------------
// 128x64 tile, 256 threads = 8 waves (4 in M x 2 in N); each wave owns 32x32 of C.
// Double-buffered software pipeline: the Tensor Data Mover fetches A-tile t+1 and
// global B loads for t+1 are in flight while the 4 WMMAs of tile t execute; a single
// workgroup barrier per k-step publishes the next buffers.
__global__ __launch_bounds__(256)
void gemm16_kernel(const _Float16* __restrict__ A, const _Float16* __restrict__ Bm,
                   const float* __restrict__ bias, const float* __restrict__ res,
                   float* __restrict__ outF, _Float16* __restrict__ outH,
                   int M, int N, int K, int act_gelu) {
    __shared__ __align__(16) _Float16 As[2][128][40];  // [buf][m][k], 8-half pad per row
    __shared__ __align__(16) _Float16 Bs[2][64][40];   // [buf][n][k] (transposed), padded
    const int tid  = threadIdx.x;
    const int lane = tid & 31, wave = tid >> 5;
    const int wm = wave & 3, wn = wave >> 2;
    const int half = lane >> 4, nloc = lane & 15;
    const int tileM = blockIdx.y * 128, tileN = blockIdx.x * 64;

    v8f acc[2][2] = {};

    const int brow = tid >> 3, bcol = (tid & 7) * 8;   // 32 k x 64 n
#if HAVE_TDM
    // D# group0/group1 per CDNA5 ISA ch.8 (2-D f16 tensor; LDS pad: 16 DWORDs data
    // then 4 DWORDs pad -> pad_interval=3, pad_amount=3 matches the [128][40] pitch).
    const unsigned ldsA[2] = {(unsigned)(uintptr_t)&As[0][0][0],
                              (unsigned)(uintptr_t)&As[1][0][0]};
    const unsigned long long abase =
        (unsigned long long)(uintptr_t)(A + (size_t)tileM * K);
    const unsigned g1_0 = (1u << 16) | (1u << 20) | (3u << 22) | (3u << 25);
    const unsigned g1_1 = ((unsigned)K & 0xffffu) << 16;           // tensor_dim0 lo
    const unsigned g1_2 = ((unsigned)K >> 16) | (((unsigned)M & 0xffffu) << 16);
    const unsigned g1_3 = ((unsigned)M >> 16) | (32u << 16);       // tile_dim0 = 32
    const unsigned g1_4 = 128u;                                    // tile_dim1 = 128
    const unsigned g1_5 = (unsigned)K;                             // dim0_stride lo
    const i32x8 G1 = {(int)g1_0, (int)g1_1, (int)g1_2, (int)g1_3,
                      (int)g1_4, (int)g1_5, 0, 0};
    const i32x4 Z4 = {0, 0, 0, 0};
    const i32x8 Z8 = {0, 0, 0, 0, 0, 0, 0, 0};
    auto tdm_issue = [&](int kk, unsigned lds) {
        const unsigned long long ga = abase + (unsigned long long)kk * 2ull;
        const u32x4 G0 = {1u,                                   // count = 1
                          lds,                                  // lds_addr
                          (unsigned)ga,                         // global_addr[31:0]
                          (unsigned)(ga >> 32) | (2u << 30)};   // addr[56:32] | type=2
        TDM_LOAD(G0, G1, Z4, Z4, Z8);
    };
#else
    const int arow = tid >> 1, acol = (tid & 1) * 16;  // 128 rows x 32 k
    v8h aR0, aR1;
#endif
    v8h bR;

    // ---- prologue: stage tile 0 ----
#if HAVE_TDM
    if (wave == 0) tdm_issue(0, ldsA[0]);
#else
    {
        const _Float16* ap = A + (size_t)(tileM + arow) * K + acol;
        aR0 = *(const v8h*)ap;
        aR1 = *(const v8h*)(ap + 8);
        *(v8h*)&As[0][arow][acol]     = aR0;
        *(v8h*)&As[0][arow][acol + 8] = aR1;
    }
#endif
    bR = *(const v8h*)(Bm + (size_t)brow * N + tileN + bcol);
    #pragma unroll
    for (int j = 0; j < 8; ++j) Bs[0][bcol + j][brow] = bR[j];
#if HAVE_TDM
    if (wave == 0) __builtin_amdgcn_s_wait_tensorcnt(0);
#endif
    __syncthreads();
    asm volatile("" ::: "memory");   // TDM wrote As outside compiler's view

    const int nT = K >> 5;
    for (int t = 0; t < nT; ++t) {
        const int cur = t & 1, nxt = cur ^ 1;
        const int k1 = (t + 1) << 5;
        const bool more = (k1 < K);
        if (more) {
            // kick off tile t+1 (DMA + VMEM) before computing tile t
#if HAVE_TDM
            if (wave == 0) tdm_issue(k1, ldsA[nxt]);
#else
            const _Float16* ap = A + (size_t)(tileM + arow) * K + k1 + acol;
            aR0 = *(const v8h*)ap;
            aR1 = *(const v8h*)(ap + 8);
#endif
            bR = *(const v8h*)(Bm + (size_t)(k1 + brow) * N + tileN + bcol);
        }

        // ---- compute tile t ----
        AFrag a[2];
        #pragma unroll
        for (int sub = 0; sub < 2; ++sub) {
            a[sub].h[0] = *(const v8h*)&As[cur][wm * 32 + sub * 16 + nloc][half * 8];
            a[sub].h[1] = *(const v8h*)&As[cur][wm * 32 + sub * 16 + nloc][16 + half * 8];
        }
        #pragma unroll
        for (int f = 0; f < 2; ++f) {
            const int col = wn * 32 + f * 16 + nloc;
            AFrag b;
            b.h[0] = *(const v8h*)&Bs[cur][col][half * 8];
            b.h[1] = *(const v8h*)&Bs[cur][col][16 + half * 8];
            #pragma unroll
            for (int sub = 0; sub < 2; ++sub)
                acc[sub][f] = __builtin_amdgcn_wmma_f32_16x16x32_f16(
                    false, a[sub].v, false, b.v, (short)0, acc[sub][f], false, false);
        }

        if (more) {
#if !HAVE_TDM
            *(v8h*)&As[nxt][arow][acol]     = aR0;
            *(v8h*)&As[nxt][arow][acol + 8] = aR1;
#endif
            #pragma unroll
            for (int j = 0; j < 8; ++j) Bs[nxt][bcol + j][brow] = bR[j];
#if HAVE_TDM
            if (wave == 0) __builtin_amdgcn_s_wait_tensorcnt(0);
#endif
        }
        __syncthreads();                 // single barrier per k-step
        asm volatile("" ::: "memory");
    }

    #pragma unroll
    for (int f = 0; f < 2; ++f) {
        const int gN = tileN + wn * 32 + f * 16 + nloc;
        const float bval = bias ? bias[gN] : 0.0f;
        #pragma unroll
        for (int sub = 0; sub < 2; ++sub) {
            #pragma unroll
            for (int i = 0; i < 8; ++i) {
                const int gM = tileM + wm * 32 + sub * 16 + i + 8 * half;
                float v = acc[sub][f][i] + bval;
                if (act_gelu) v = gelu_f(v);
                if (res)  v += res[(size_t)gM * N + gN];
                if (outF) outF[(size_t)gM * N + gN] = v;
                if (outH) outH[(size_t)gM * N + gN] = (_Float16)v;
            }
        }
    }
}

// ---------------- split qkv (f16, [B*S,3D]) into per-head [B,H,S,HD] ----------------
__global__ void pack_qkv_kernel(const _Float16* __restrict__ qkv,
                                _Float16* __restrict__ q, _Float16* __restrict__ k,
                                _Float16* __restrict__ v) {
    size_t i = (size_t)blockIdx.x * blockDim.x + threadIdx.x;
    if (i >= (size_t)B_ * S_ * D_) return;
    const int hd = (int)(i % HD_);
    size_t t = i / HD_;
    const int hh = (int)(t % H_);
    const size_t bs = t / H_;
    const size_t b = bs / S_, s = bs % S_;
    const size_t src = bs * (size_t)(3 * D_) + (size_t)hh * HD_ + hd;
    const size_t dst = (((b * H_) + hh) * S_ + s) * HD_ + hd;
    q[dst] = qkv[src];
    k[dst] = qkv[src + D_];
    v[dst] = qkv[src + 2 * D_];
}

// ---------------- flash-style attention, faithful rel-bias/mask semantics ----------------
// Reference: w = (qk*mask - 1e10*(1-mask)) * (relw*mask)  ==>  masked logits become 0
// (NOT -inf), so every key tile participates with s=0 on masked entries.
// K tile staged by the Tensor Data Mover (single wave per block: TDM ignores EXEC),
// overlapping the manual V transpose staging.
__global__ __launch_bounds__(32)
void attn_kernel(const _Float16* __restrict__ qg, const _Float16* __restrict__ kg,
                 const _Float16* __restrict__ vg, const int* __restrict__ rel,
                 const float* __restrict__ rel_emb, _Float16* __restrict__ outH) {
    __shared__ __align__(16) _Float16 Kt[32][72];  // [key][hd], 128B data + 16B pad
    __shared__ __align__(16) _Float16 Vt[64][40];  // transposed: [hd][key], padded
    __shared__ __align__(16) _Float16 Ps[16][40];  // [qrow][key], padded
    const int lane = threadIdx.x;
    const int half = lane >> 4, nloc = lane & 15;
    const int nqt = S_ / 16;
    const int qt = blockIdx.x % nqt;
    const int hh = (blockIdx.x / nqt) % H_;
    const int b  = blockIdx.x / (nqt * H_);
    const int qbase = qt * 16;
    const size_t headBase = ((size_t)(b * H_ + hh)) * S_ * HD_;
    const float invs = 0.125f;  // 1/sqrt(HD)

    AFrag qf[2];
    #pragma unroll
    for (int c = 0; c < 2; ++c) {
        const _Float16* p = qg + headBase + (size_t)(qbase + nloc) * HD_ + c * 32 + half * 8;
        qf[c].h[0] = *(const v8h*)p;
        qf[c].h[1] = *(const v8h*)(p + 16);
    }

#if HAVE_TDM
    // D# for one K tile: 32 rows x 64 f16, row stride HD; LDS pad: 32 DWORDs data
    // then 4 DWORDs pad (interval code 4, amount code 3) -> matches Kt[32][72].
    const unsigned ldsK = (unsigned)(uintptr_t)&Kt[0][0];
    const unsigned long long kbaseAddr =
        (unsigned long long)(uintptr_t)(kg + headBase);
    const unsigned kg1_0 = (1u << 16) | (1u << 20) | (4u << 22) | (3u << 25);
    const unsigned kg1_1 = 64u << 16;                        // tensor_dim0 = 64 (lo)
    const unsigned kg1_2 = (((unsigned)S_ & 0xffffu) << 16); // dim0 hi=0 | dim1 lo
    const unsigned kg1_3 = (64u << 16);                      // dim1 hi=0 | tile_dim0=64
    const unsigned kg1_4 = 32u;                              // tile_dim1 = 32
    const unsigned kg1_5 = 64u;                              // dim0_stride = 64
    const i32x8 KG1 = {(int)kg1_0, (int)kg1_1, (int)kg1_2, (int)kg1_3,
                       (int)kg1_4, (int)kg1_5, 0, 0};
    const i32x4 Z4 = {0, 0, 0, 0};
    const i32x8 Z8 = {0, 0, 0, 0, 0, 0, 0, 0};
#endif

    v8f acc[4] = {};
    float mrow[8], lrow[8];
    #pragma unroll
    for (int i = 0; i < 8; ++i) { mrow[i] = -1e30f; lrow[i] = 0.0f; }

    for (int kb = 0; kb < S_; kb += 32) {
        // prefetch next tile's rel rows toward L2 (dominant gather stream)
        if (kb + 32 < S_)
            __builtin_prefetch(
                &rel[((size_t)b * S_ + qbase + nloc) * S_ + kb + 32 + half * 16], 0, 1);

#if HAVE_TDM
        {   // TDM fetch of the K tile; overlaps the manual V staging below
            const unsigned long long ga =
                kbaseAddr + (unsigned long long)kb * (HD_ * 2ull);
            const u32x4 G0 = {1u, ldsK, (unsigned)ga,
                              (unsigned)(ga >> 32) | (2u << 30)};
            TDM_LOAD(G0, KG1, Z4, Z4, Z8);
        }
        const _Float16* vp = vg + headBase + (size_t)(kb + lane) * HD_;
        #pragma unroll
        for (int j = 0; j < 8; ++j) {
            v8h vv = *(const v8h*)(vp + j * 8);
            #pragma unroll
            for (int e = 0; e < 8; ++e) Vt[j * 8 + e][lane] = vv[e];
        }
        __builtin_amdgcn_s_wait_tensorcnt(0);
        asm volatile("" ::: "memory");   // TDM wrote Kt outside compiler's view
#else
        const _Float16* kp = kg + headBase + (size_t)(kb + lane) * HD_;
        const _Float16* vp = vg + headBase + (size_t)(kb + lane) * HD_;
        #pragma unroll
        for (int j = 0; j < 8; ++j) {
            v8h kv = *(const v8h*)(kp + j * 8);
            *(v8h*)&Kt[lane][j * 8] = kv;
            v8h vv = *(const v8h*)(vp + j * 8);
            #pragma unroll
            for (int e = 0; e < 8; ++e) Vt[j * 8 + e][lane] = vv[e];
        }
#endif
        // single wave: LDS ops are in-order within the wave; compiler inserts waits.

        // scores: Q(16x64) @ K^T(64x32) as 2 chunks x 2 key-fragments
        v8f sc[2] = {};
        #pragma unroll
        for (int c = 0; c < 2; ++c) {
            #pragma unroll
            for (int f = 0; f < 2; ++f) {
                AFrag bk;
                bk.h[0] = *(const v8h*)&Kt[f * 16 + nloc][c * 32 + half * 8];
                bk.h[1] = *(const v8h*)&Kt[f * 16 + nloc][c * 32 + 16 + half * 8];
                sc[f] = __builtin_amdgcn_wmma_f32_16x16x32_f16(
                    false, qf[c].v, false, bk.v, (short)0, sc[f], false, false);
            }
        }

        // mask + rel-bias multiply + online softmax (row stats across 16-lane half)
        #pragma unroll
        for (int i = 0; i < 8; ++i) {
            const int qrow = qbase + i + 8 * half;
            float s0 = 0.0f, s1 = 0.0f;
            int key = kb + nloc;
            if (key <= qrow) {
                const int idx = rel[((size_t)b * S_ + qrow) * S_ + key];
                s0 = sc[0][i] * invs * rel_emb[idx * H_ + hh];
            }
            key = kb + 16 + nloc;
            if (key <= qrow) {
                const int idx = rel[((size_t)b * S_ + qrow) * S_ + key];
                s1 = sc[1][i] * invs * rel_emb[idx * H_ + hh];
            }
            float t = fmaxf(s0, s1);
            #pragma unroll
            for (int off = 8; off > 0; off >>= 1) t = fmaxf(t, __shfl_xor(t, off, 32));
            const float mnew  = fmaxf(mrow[i], t);
            const float scale = __expf(mrow[i] - mnew);
            const float p0 = __expf(s0 - mnew);
            const float p1 = __expf(s1 - mnew);
            float srow = p0 + p1;
            #pragma unroll
            for (int off = 8; off > 0; off >>= 1) srow += __shfl_xor(srow, off, 32);
            lrow[i] = lrow[i] * scale + srow;
            mrow[i] = mnew;
            #pragma unroll
            for (int j = 0; j < 4; ++j) acc[j][i] *= scale;
            Ps[i + 8 * half][nloc]      = (_Float16)p0;
            Ps[i + 8 * half][16 + nloc] = (_Float16)p1;
        }

        // P(16x32) @ V(32x64) -> 4 accumulators
        AFrag ap;
        ap.h[0] = *(const v8h*)&Ps[nloc][half * 8];
        ap.h[1] = *(const v8h*)&Ps[nloc][16 + half * 8];
        #pragma unroll
        for (int j = 0; j < 4; ++j) {
            AFrag bv;
            bv.h[0] = *(const v8h*)&Vt[j * 16 + nloc][half * 8];
            bv.h[1] = *(const v8h*)&Vt[j * 16 + nloc][16 + half * 8];
            acc[j] = __builtin_amdgcn_wmma_f32_16x16x32_f16(
                false, ap.v, false, bv.v, (short)0, acc[j], false, false);
        }
    }

    // epilogue: normalize and write merged-head f16 [B*S, D]
    #pragma unroll
    for (int j = 0; j < 4; ++j) {
        const int hd = j * 16 + nloc;
        #pragma unroll
        for (int i = 0; i < 8; ++i) {
            const int qrow = qbase + i + 8 * half;
            const float ov = acc[j][i] / lrow[i];
            outH[((size_t)(b * S_ + qrow)) * D_ + hh * HD_ + hd] = (_Float16)ov;
        }
    }
}

// ---------------- host-side orchestration ----------------
extern "C" void kernel_launch(void* const* d_in, const int* in_sizes, int n_in,
                              void* d_out, int out_size, void* d_ws, size_t ws_size,
                              hipStream_t stream) {
    (void)in_sizes; (void)n_in; (void)out_size; (void)ws_size;
    const float* x       = (const float*)d_in[0];
    const int*   rel     = (const int*)  d_in[1];
    const float* ln1_w   = (const float*)d_in[2];
    const float* ln1_b   = (const float*)d_in[3];
    const float* Wqkv    = (const float*)d_in[4];
    const float* bqkv    = (const float*)d_in[5];
    const float* Wo      = (const float*)d_in[6];
    const float* bo      = (const float*)d_in[7];
    const float* rel_emb = (const float*)d_in[8];
    const float* ln2_w   = (const float*)d_in[9];
    const float* ln2_b   = (const float*)d_in[10];
    const float* Wfc     = (const float*)d_in[11];
    const float* bfc     = (const float*)d_in[12];
    const float* Wp      = (const float*)d_in[13];
    const float* bp      = (const float*)d_in[14];

    char* ws = (char*)d_ws;
    size_t off = 0;
    auto alloc = [&](size_t bytes) -> char* {
        char* p = ws + off;
        off += (bytes + 255) & ~(size_t)255;
        return p;
    };
    const size_t MS = (size_t)B_ * S_;                 // 4096 rows
    _Float16* h1     = (_Float16*)alloc(MS * D_ * 2);
    _Float16* qkv16  = (_Float16*)alloc(MS * 3 * D_ * 2);
    _Float16* q16    = (_Float16*)alloc(MS * D_ * 2);
    _Float16* k16    = (_Float16*)alloc(MS * D_ * 2);
    _Float16* v16    = (_Float16*)alloc(MS * D_ * 2);
    _Float16* ao16   = (_Float16*)alloc(MS * D_ * 2);
    float*    x1     = (float*)   alloc(MS * D_ * 4);
    _Float16* h2     = (_Float16*)alloc(MS * D_ * 2);
    _Float16* fc16   = (_Float16*)alloc(MS * 4 * D_ * 2);
    _Float16* wqkv16 = (_Float16*)alloc((size_t)D_ * 3 * D_ * 2);
    _Float16* wo16   = (_Float16*)alloc((size_t)D_ * D_ * 2);
    _Float16* wfc16  = (_Float16*)alloc((size_t)D_ * 4 * D_ * 2);
    _Float16* wp16   = (_Float16*)alloc((size_t)4 * D_ * D_ * 2);

    auto cvt = [&](const float* src, _Float16* dst, size_t n) {
        cvt_f32_f16_kernel<<<(unsigned)((n + 255) / 256), 256, 0, stream>>>(src, dst, n);
    };
    cvt(Wqkv, wqkv16, (size_t)D_ * 3 * D_);
    cvt(Wo,   wo16,   (size_t)D_ * D_);
    cvt(Wfc,  wfc16,  (size_t)D_ * 4 * D_);
    cvt(Wp,   wp16,   (size_t)4 * D_ * D_);

    // h = LN1(x)
    layernorm_kernel<<<(unsigned)MS, 256, 0, stream>>>(x, ln1_w, ln1_b, h1);

    // qkv = h @ Wqkv + bqkv   (f16 out)
    gemm16_kernel<<<dim3(3 * D_ / 64, (unsigned)(MS / 128)), 256, 0, stream>>>(
        h1, wqkv16, bqkv, nullptr, nullptr, qkv16, (int)MS, 3 * D_, D_, 0);

    // split heads
    pack_qkv_kernel<<<(unsigned)((MS * D_ + 255) / 256), 256, 0, stream>>>(
        qkv16, q16, k16, v16);

    // attention (flash-style, faithful rel*mask semantics), merged-head f16 out
    attn_kernel<<<(unsigned)(B_ * H_ * (S_ / 16)), 32, 0, stream>>>(
        q16, k16, v16, rel, rel_emb, ao16);

    // x1 = x + attn @ Wo + bo   (f32 out)
    gemm16_kernel<<<dim3(D_ / 64, (unsigned)(MS / 128)), 256, 0, stream>>>(
        ao16, wo16, bo, x, x1, nullptr, (int)MS, D_, D_, 0);

    // h2 = LN2(x1)
    layernorm_kernel<<<(unsigned)MS, 256, 0, stream>>>(x1, ln2_w, ln2_b, h2);

    // fc = gelu(h2 @ Wfc + bfc)   (f16 out)
    gemm16_kernel<<<dim3(4 * D_ / 64, (unsigned)(MS / 128)), 256, 0, stream>>>(
        h2, wfc16, bfc, nullptr, nullptr, fc16, (int)MS, 4 * D_, D_, 1);

    // out = x1 + fc @ Wp + bp   (f32 out -> d_out)
    gemm16_kernel<<<dim3(D_ / 64, (unsigned)(MS / 128)), 256, 0, stream>>>(
        fc16, wp16, bp, x1, (float*)d_out, nullptr, (int)MS, D_, 4 * D_, 0);
}
